// VQLayer_86320252715229
// MI455X (gfx1250) — matrix operand order repaired and checked by reading, not compiled, not run on hardware
//
#include <hip/hip_runtime.h>

#define N_ROWS  65536
#define K_PROT  1024
#define D_DIM   256
#define ROWS_PB 64          // rows per block (4 waves x 16 rows)
#define ESTRIDE 1028        // padded LDS row stride (floats): halves hit disjoint banks

typedef __attribute__((ext_vector_type(16))) __bf16 v16bf;
typedef __attribute__((ext_vector_type(8)))  float  v8f;

union V16 { unsigned int u[8]; v16bf v; };

__device__ __forceinline__ unsigned short f2bf(float f) {
  unsigned u = __builtin_bit_cast(unsigned, f);
  u += 0x7FFFu + ((u >> 16) & 1u);            // round-to-nearest-even
  return (unsigned short)(u >> 16);
}
__device__ __forceinline__ unsigned pack2(float a, float b) {
  return (unsigned)f2bf(a) | ((unsigned)f2bf(b) << 16);
}

// ---------------- zero accumulators ----------------
__global__ void vq_init_kernel(float* __restrict__ accum, float* __restrict__ sqsum) {
  int t = blockIdx.x * 256 + threadIdx.x;
  if (t < K_PROT) accum[t] = 0.f;
  if (t == 0) sqsum[0] = 0.f;
}

// ---------------- protos -> bf16 B-fragment layout ----------------
// short index = ((jt*8 + ks)*32 + lane)*16 + e  holds bf16(P[jt*16+(lane&15)][ks*32 + ((e&8)<<1) + ((lane>>4)<<3) + (e&7)])
__global__ void vq_pack_kernel(const float* __restrict__ protos,
                               unsigned short* __restrict__ packB) {
  int idx = blockIdx.x * 256 + threadIdx.x;           // 64*8*32*16 = 262144
  int e    = idx & 15;
  int lane = (idx >> 4) & 31;
  int ks   = (idx >> 9) & 7;
  int jt   = idx >> 12;
  int j = jt * 16 + (lane & 15);
  int d = ks * 32 + ((e & 8) << 1) + ((lane >> 4) << 3) + (e & 7);
  packB[idx] = f2bf(protos[j * D_DIM + d]);
}

// ---------------- ||proto_j||^2 ----------------
__global__ void vq_pb2_kernel(const float* __restrict__ protos, float* __restrict__ pb2) {
  int j = blockIdx.x * 256 + threadIdx.x;
  if (j < K_PROT) {
    float s = 0.f;
    for (int d = 0; d < D_DIM; ++d) { float x = protos[j * D_DIM + d]; s += x * x; }
    pb2[j] = s;
  }
}

// ---------------- fused main kernel ----------------
__global__ __launch_bounds__(128)
void vq_main_kernel(const float* __restrict__ lat,
                    const float* __restrict__ mus,
                    const float* __restrict__ protos,
                    const unsigned int* __restrict__ packB,
                    const float* __restrict__ pb2,
                    float* __restrict__ out,
                    float* __restrict__ accum,
                    float* __restrict__ sqsum)
{
  extern __shared__ float smemF[];
  float* exps    = smemF;                         // ROWS_PB * ESTRIDE
  float* pb2s    = smemF + ROWS_PB * ESTRIDE;     // K_PROT
  float* rowMu2s = pb2s + K_PROT;                 // ROWS_PB
  float* rowInv  = rowMu2s + ROWS_PB;             // ROWS_PB
  int*   rowIdx  = (int*)(rowInv + ROWS_PB);      // ROWS_PB

  const int tid  = threadIdx.x;
  const int lane = tid & 31;
  const int wave = tid >> 5;
  const int wRow = wave * 16;
  const long rowBase = (long)blockIdx.x * ROWS_PB;

  for (int j = tid; j < K_PROT; j += 128) pb2s[j] = pb2[j];

  // ---- build bf16 A fragments (16x32 per k-step) for latents & mus ----
  // 16-bit A layout: lanes 0-15 (row = lane) hold d-offsets {0..7, 16..23},
  // lanes 16-31 (same rows) hold {8..15, 24..31} relative to ks*32.
  V16 aL[8], aM[8];
  float mu2 = 0.f;
  {
    const long r = rowBase + wRow + (lane & 15);
    const float* lrow = lat + r * D_DIM;
    const float* mrow = mus + r * D_DIM;
    const int off = (lane < 16) ? 0 : 8;
#pragma unroll
    for (int ks = 0; ks < 8; ++ks) {
      const int d0 = ks * 32 + off;
      float4 l0 = *(const float4*)(lrow + d0);
      float4 l1 = *(const float4*)(lrow + d0 + 4);
      float4 l2 = *(const float4*)(lrow + d0 + 16);
      float4 l3 = *(const float4*)(lrow + d0 + 20);
      aL[ks].u[0] = pack2(l0.x, l0.y); aL[ks].u[1] = pack2(l0.z, l0.w);
      aL[ks].u[2] = pack2(l1.x, l1.y); aL[ks].u[3] = pack2(l1.z, l1.w);
      aL[ks].u[4] = pack2(l2.x, l2.y); aL[ks].u[5] = pack2(l2.z, l2.w);
      aL[ks].u[6] = pack2(l3.x, l3.y); aL[ks].u[7] = pack2(l3.z, l3.w);
      float4 m0 = *(const float4*)(mrow + d0);
      float4 m1 = *(const float4*)(mrow + d0 + 4);
      float4 m2 = *(const float4*)(mrow + d0 + 16);
      float4 m3 = *(const float4*)(mrow + d0 + 20);
      aM[ks].u[0] = pack2(m0.x, m0.y); aM[ks].u[1] = pack2(m0.z, m0.w);
      aM[ks].u[2] = pack2(m1.x, m1.y); aM[ks].u[3] = pack2(m1.z, m1.w);
      aM[ks].u[4] = pack2(m2.x, m2.y); aM[ks].u[5] = pack2(m2.z, m2.w);
      aM[ks].u[6] = pack2(m3.x, m3.y); aM[ks].u[7] = pack2(m3.z, m3.w);
      mu2 += m0.x*m0.x + m0.y*m0.y + m0.z*m0.z + m0.w*m0.w
           + m1.x*m1.x + m1.y*m1.y + m1.z*m1.z + m1.w*m1.w
           + m2.x*m2.x + m2.y*m2.y + m2.z*m2.z + m2.w*m2.w
           + m3.x*m3.x + m3.y*m3.y + m3.z*m3.z + m3.w*m3.w;
    }
  }
  mu2 += __shfl_xor(mu2, 16, 32);               // lanes r and r+16 together cover D
  if (lane < 16) rowMu2s[wRow + lane] = mu2;
  __syncthreads();

  const int col = lane & 15;
  const int hf  = lane >> 4;
  float rM2[8];
#pragma unroll
  for (int v = 0; v < 8; ++v) rM2[v] = rowMu2s[wRow + v + 8 * hf];

  float minV[8]; int minI[8]; float eSum[8];
#pragma unroll
  for (int v = 0; v < 8; ++v) { minV[v] = 3.4e38f; minI[v] = 0; eSum[v] = 0.f; }

  // ---- sweep all 64 column tiles of 16 prototypes ----
  for (int jt = 0; jt < K_PROT / 16; ++jt) {
    const unsigned int* bp = packB + ((size_t)jt * 8 * 32 + lane) * 8;
    if (jt + 1 < K_PROT / 16)
      __builtin_prefetch(packB + ((size_t)(jt + 1) * 8 * 32) * 8, 0, 0);
    v8f accL = {}; v8f accM = {};
#pragma unroll
    for (int ks = 0; ks < 8; ++ks) {
      V16 b;
      const uint4* q = (const uint4*)(bp + (size_t)ks * 32 * 8);
      uint4 q0 = q[0]; uint4 q1 = q[1];
      b.u[0] = q0.x; b.u[1] = q0.y; b.u[2] = q0.z; b.u[3] = q0.w;
      b.u[4] = q1.x; b.u[5] = q1.y; b.u[6] = q1.z; b.u[7] = q1.w;
      accL = __builtin_amdgcn_wmma_f32_16x16x32_bf16(false, aL[ks].v, false, b.v,
                                                     (short)0, accL, false, false);
      accM = __builtin_amdgcn_wmma_f32_16x16x32_bf16(false, aM[ks].v, false, b.v,
                                                     (short)0, accM, false, false);
    }
    const float pb = pb2s[jt * 16 + col];
#pragma unroll
    for (int v = 0; v < 8; ++v) {
      const int rloc = wRow + v + 8 * hf;       // C layout: row = v + 8*(lane>=16)
      const float sL = pb - 2.0f * accL[v];     // ||l||^2 constant per row -> drop for argmin
      if (sL < minV[v]) { minV[v] = sL; minI[v] = jt * 16 + col; }
      const float dM = rM2[v] + pb - 2.0f * accM[v];
      const float e = __expf(-dM) + 1e-8f;
      exps[rloc * ESTRIDE + jt * 16 + col] = e;
      eSum[v] += e;
    }
  }

  // ---- cross-lane reductions within each 16-lane row group ----
#pragma unroll
  for (int v = 0; v < 8; ++v) {
    float mv = minV[v]; int mi = minI[v];
#pragma unroll
    for (int m = 8; m >= 1; m >>= 1) {
      const float ov = __shfl_xor(mv, m, 32);
      const int   oi = __shfl_xor(mi, m, 32);
      if (ov < mv || (ov == mv && oi < mi)) { mv = ov; mi = oi; }
    }
    float es = eSum[v];
#pragma unroll
    for (int m = 8; m >= 1; m >>= 1) es += __shfl_xor(es, m, 32);
    if (col == 0) {
      rowIdx[wRow + v + 8 * hf] = mi;
      rowInv[wRow + v + 8 * hf] = 1.0f / es;
    }
  }
  __syncthreads();

  // ---- entropy numerator: one atomic per k per block ----
  for (int j = tid; j < K_PROT; j += 128) {
    float s = 0.f;
#pragma unroll 8
    for (int r = 0; r < ROWS_PB; ++r) s += exps[r * ESTRIDE + j] * rowInv[r];
    atomicAdd(&accum[j], s);
  }

  // ---- gather quantized rows + (quantized - mus)^2 sum ----
  float sq = 0.f;
  for (int r = 0; r < 16; ++r) {
    const long gr = rowBase + wRow + r;
    const int  ci = rowIdx[wRow + r];
    const float4* ps = (const float4*)(protos + (size_t)ci * D_DIM) + lane * 2;
    const float4* ms = (const float4*)(mus + gr * D_DIM) + lane * 2;
    float4* os = (float4*)(out + gr * D_DIM) + lane * 2;
    float4 p0 = ps[0], p1 = ps[1];
    float4 m0 = ms[0], m1 = ms[1];
    os[0] = p0; os[1] = p1;
    float dx;
    dx = p0.x - m0.x; sq += dx*dx;  dx = p0.y - m0.y; sq += dx*dx;
    dx = p0.z - m0.z; sq += dx*dx;  dx = p0.w - m0.w; sq += dx*dx;
    dx = p1.x - m1.x; sq += dx*dx;  dx = p1.y - m1.y; sq += dx*dx;
    dx = p1.z - m1.z; sq += dx*dx;  dx = p1.w - m1.w; sq += dx*dx;
  }
#pragma unroll
  for (int m = 16; m >= 1; m >>= 1) sq += __shfl_xor(sq, m, 32);
  if (lane == 0) atomicAdd(sqsum, sq);
}

// ---------------- entropy + loss finalize ----------------
__global__ __launch_bounds__(1024)
void vq_finalize_kernel(const float* __restrict__ accum,
                        const float* __restrict__ sqsum,
                        float* __restrict__ lossOut) {
  __shared__ float red[1024];
  const int t = threadIdx.x;
  float p = accum[t] * (1.0f / (float)N_ROWS);
  red[t] = (p > 0.f) ? (-p * __logf(p)) : 0.f;
  __syncthreads();
  for (int s = 512; s > 0; s >>= 1) {
    if (t < s) red[t] += red[t + s];
    __syncthreads();
  }
  if (t == 0) {
    const float mse = sqsum[0] / ((float)N_ROWS * (float)D_DIM);
    lossOut[0] = 1.25f * mse + 0.01f * red[0];   // (beta+1)*mse + entropy_weight*ent
  }
}

extern "C" void kernel_launch(void* const* d_in, const int* in_sizes, int n_in,
                              void* d_out, int out_size, void* d_ws, size_t ws_size,
                              hipStream_t stream) {
  const float* lat    = (const float*)d_in[0];   // [65536,256]
  const float* mus    = (const float*)d_in[1];   // [65536,256]
  const float* protos = (const float*)d_in[2];   // [1024,256]
  float* out = (float*)d_out;                    // [65536*256] quantized + [1] loss

  char* ws = (char*)d_ws;
  unsigned short* packB = (unsigned short*)ws;             // 524288 B
  float* pb2   = (float*)(ws + 524288);                    // 4096 B
  float* accum = (float*)(ws + 524288 + 4096);             // 4096 B
  float* sqsum = (float*)(ws + 524288 + 8192);             // 16 B

  vq_init_kernel<<<4, 256, 0, stream>>>(accum, sqsum);
  vq_pack_kernel<<<1024, 256, 0, stream>>>(protos, packB);
  vq_pb2_kernel<<<4, 256, 0, stream>>>(protos, pb2);

  const size_t smem = (size_t)(ROWS_PB * ESTRIDE + K_PROT + 3 * ROWS_PB) * sizeof(float);
  vq_main_kernel<<<N_ROWS / ROWS_PB, 128, smem, stream>>>(
      lat, mus, protos, (const unsigned int*)packB, pb2, out, accum, sqsum);

  vq_finalize_kernel<<<1, 1024, 0, stream>>>(accum, sqsum, out + (size_t)N_ROWS * D_DIM);
}
